// QuantumMultiHeadAttention_21534966022686
// MI455X (gfx1250) — compile-verified, matrix operand order checked by reading, not run on hardware
//
#include <hip/hip_runtime.h>
#include <math.h>

// ---------------- types / constants ----------------
typedef __bf16 bf16;
typedef __attribute__((ext_vector_type(16))) __bf16 v16bf;
typedef __attribute__((ext_vector_type(8)))  float  v8f;
typedef __attribute__((ext_vector_type(4)))  unsigned int u32x4;
typedef __attribute__((ext_vector_type(8)))  int i32x8;
typedef __attribute__((ext_vector_type(4)))  int i32x4;

#define BB   2
#define SL   1024
#define DD   1024
#define NH   16
#define DKK  64
#define BHS  (BB * NH * SL)          // 32768 rows of per-head Q/K
#define MTOT (BB * SL)               // 2048
#define SMLD 272                     // LDS row pitch in bf16 elems (256 + 16 pad)

// ---------------- WMMA fragment helpers ----------------
// 16-bit A/B fragment layout (CDNA5 ISA 7.12.2):
//   lanes 0-15 : row M = lane,    VGPR0-3 = K 0..7,   VGPR4-7 = K 16..23
//   lanes 16-31: row M = lane-16, VGPR0-3 = K 8..15,  VGPR4-7 = K 24..31
static __device__ __forceinline__ v16bf load_frag_bf16(const bf16* base, int ld, int k0) {
    const int lane  = threadIdx.x & 31;
    const int row   = lane & 15;
    const int khalf = (lane >> 4) << 3;            // 0 or 8
    const bf16* p = base + row * ld + k0 + khalf;
    union { v16bf v; uint4 u[2]; } t;
    t.u[0] = *reinterpret_cast<const uint4*>(p);        // K khalf .. khalf+7
    t.u[1] = *reinterpret_cast<const uint4*>(p + 16);   // K khalf+16 .. khalf+23
    return t.v;
}

// f32 source (attn) -> bf16 fragment on the fly.
static __device__ __forceinline__ v16bf load_frag_f32(const float* base, int ld, int k0) {
    const int lane  = threadIdx.x & 31;
    const int row   = lane & 15;
    const int khalf = (lane >> 4) << 3;
    const float* p = base + row * ld + k0 + khalf;
    float4 a = *reinterpret_cast<const float4*>(p);
    float4 b = *reinterpret_cast<const float4*>(p + 4);
    float4 c = *reinterpret_cast<const float4*>(p + 16);
    float4 d = *reinterpret_cast<const float4*>(p + 20);
    v16bf f;
    f[0]=(bf16)a.x;  f[1]=(bf16)a.y;  f[2]=(bf16)a.z;  f[3]=(bf16)a.w;
    f[4]=(bf16)b.x;  f[5]=(bf16)b.y;  f[6]=(bf16)b.z;  f[7]=(bf16)b.w;
    f[8]=(bf16)c.x;  f[9]=(bf16)c.y;  f[10]=(bf16)c.z; f[11]=(bf16)c.w;
    f[12]=(bf16)d.x; f[13]=(bf16)d.y; f[14]=(bf16)d.z; f[15]=(bf16)d.w;
    return f;
}

// Fragment read from an LDS-staged panel (emits ds_load_b128 pairs).
static __device__ __forceinline__ v16bf load_frag_lds(const bf16* sm, int row0, int k) {
    const int lane  = threadIdx.x & 31;
    const int row   = row0 + (lane & 15);
    const int khalf = (lane >> 4) << 3;
    const bf16* p = sm + row * SMLD + k + khalf;
    union { v16bf v; uint4 u[2]; } t;
    t.u[0] = *reinterpret_cast<const uint4*>(p);
    t.u[1] = *reinterpret_cast<const uint4*>(p + 16);
    return t.v;
}

static __device__ __forceinline__ v8f wmma_bf16(v16bf a, v16bf b, v8f c) {
    return __builtin_amdgcn_wmma_f32_16x16x32_bf16(false, a, false, b, (short)0, c, false, false);
}

// ---------------- Tensor Data Mover: stage a 64-row x 256-col bf16 panel ----------------
// D# per CDNA5 ISA. data_size=2B, 2D tile, pad_enable with
// pad_interval=128 DW (one 256-elem row) and pad_amount=8 DW (16 elems) so the
// LDS pitch is SMLD=272 elems -> fragment rows hit different LDS banks.
static __device__ __forceinline__ void tdm_stage_panel(unsigned lds_off,
                                                       unsigned long long gaddr,
                                                       unsigned tensor_stride_elems) {
    u32x4 g0;
    g0.x = 1u;                                             // count=1 (user descriptor)
    g0.y = lds_off;                                        // LDS byte address
    g0.z = (unsigned)(gaddr & 0xFFFFFFFFull);              // global_addr[31:0]
    g0.w = (unsigned)((gaddr >> 32) & 0x01FFFFFFull) | (2u << 30);   // addr[56:32] | type=2
    i32x8 g1;
    g1[0] = (int)((1u << 16)            // data_size = 1 -> 2 bytes
                | (1u << 20)            // pad_enable
                | (6u << 22)            // pad_interval: 128 DWORDs (one row)
                | (7u << 25));          // pad_amount: 8 DWORDs (16 bf16)
    g1[1] = (int)(tensor_stride_elems << 16);   // tensor_dim0 low16 @ bits 63:48
    g1[2] = (int)((tensor_stride_elems >> 16) | (64u << 16));  // dim0 hi | tensor_dim1=64
    g1[3] = (int)(256u << 16);                  // tile_dim0 = 256 elems @ bits 127:112
    g1[4] = 64;                                 // tile_dim1 = 64 rows
    g1[5] = (int)tensor_stride_elems;           // tensor_dim0_stride low 32
    g1[6] = 0;
    g1[7] = 0;
    i32x4 gz = {};
#if defined(__clang_major__) && __clang_major__ >= 23
    i32x8 gz8 = {};
    __builtin_amdgcn_tensor_load_to_lds(g0, g1, gz, gz, gz8, 0);
#else
    __builtin_amdgcn_tensor_load_to_lds(g0, g1, gz, gz, 0);
#endif
}

// ---------------- kernels ----------------
__global__ void cvt_f32_bf16(const float* __restrict__ src, bf16* __restrict__ dst, int n) {
    int i = blockIdx.x * blockDim.x + threadIdx.x;
    if (i < n) dst[i] = (bf16)src[i];
}

// QKV projection: block computes 256 rows x 64 cols; the 64-row W panel is
// TDM-staged into LDS in 4 K-chunks and shared by all 8 waves.
// mode 0: dst[b,h,s,c]  (Q, K)      mode 1: dst[b,h,c,s]  (V transposed)
__global__ void proj_qkv(const bf16* __restrict__ X, const bf16* __restrict__ W,
                         const float* __restrict__ bias, bf16* __restrict__ dst, int mode) {
    __shared__ __align__(16) bf16 smB[64 * SMLD];
    const int wave = threadIdx.x >> 5;
    const int bm = blockIdx.x >> 4;            // 8 blocks of 256 rows
    const int bn = blockIdx.x & 15;            // 16 blocks of 64 cols
    const bf16* Ab = X + ((size_t)bm * 256 + wave * 32) * DD;
    const bf16* Bg = W + (size_t)bn * 64 * DD;
    v8f acc[2][4] = {};
    for (int ch = 0; ch < 4; ++ch) {
        __syncthreads();                       // previous chunk fully consumed
        if (threadIdx.x < 32) {                // wave 0 drives the TDM
            tdm_stage_panel((unsigned)(unsigned long long)(void*)smB,
                            (unsigned long long)(const void*)(Bg + ch * 256), DD);
            __builtin_amdgcn_s_wait_tensorcnt(0);
        }
        __syncthreads();
        for (int kk = 0; kk < 256; kk += 32) {
            const int k0 = ch * 256 + kk;
            __builtin_prefetch(Ab + ((threadIdx.x & 15) * DD) + k0 + 64, 0, 0);
            v16bf a0 = load_frag_bf16(Ab, DD, k0);
            v16bf a1 = load_frag_bf16(Ab + 16 * DD, DD, k0);
#pragma unroll
            for (int j = 0; j < 4; j++) {
                v16bf b = load_frag_lds(smB, j * 16, kk);
                acc[0][j] = wmma_bf16(a0, b, acc[0][j]);
                acc[1][j] = wmma_bf16(a1, b, acc[1][j]);
            }
        }
    }
    const int lane  = threadIdx.x & 31;
    const int nloc  = lane & 15;
    const int mbase = (lane >> 4) << 3;
#pragma unroll
    for (int ms = 0; ms < 2; ms++) {
#pragma unroll
        for (int j = 0; j < 4; j++) {
            const int ncol = bn * 64 + j * 16 + nloc;
            const float bvv = bias[ncol];
            const int h = ncol >> 6, c = ncol & 63;
#pragma unroll
            for (int v = 0; v < 8; v++) {
                const int m  = bm * 256 + wave * 32 + ms * 16 + mbase + v;
                const int b_ = m >> 10, s = m & (SL - 1);
                const float val = acc[ms][j][v] + bvv;
                if (mode)
                    dst[(((size_t)(b_ * NH + h) * DKK + c) * SL) + s] = (bf16)val;
                else
                    dst[(((size_t)(b_ * NH + h) * SL + s) * DKK) + c] = (bf16)val;
            }
        }
    }
}

// r = cos(pi * sigmoid(q . Wt[e,:] + bt[e])); Q variant folds strengths[h].
__global__ void theta_r(const bf16* __restrict__ src, bf16* __restrict__ dst,
                        const float* __restrict__ Wt, const float* __restrict__ bt,
                        const float* __restrict__ strengths, int isQ) {
    const int gid = blockIdx.x * blockDim.x + threadIdx.x;
    const int row = gid >> 6;          // (b*H+h)*S + s
    const int e   = gid & 63;
    if (row >= BHS) return;
    const bf16* q = src + (size_t)row * DKK;
    const float* w = Wt + e * DKK;
    float t = bt[e];
#pragma unroll 8
    for (int d = 0; d < DKK; d++) t += (float)q[d] * w[d];
    const float sig = 1.0f / (1.0f + __expf(-t));
    float r = __cosf(3.14159265358979323846f * sig);
    if (isQ) r *= strengths[(row >> 10) & (NH - 1)];
    dst[(size_t)row * DKK + e] = (bf16)r;
}

// quantum[b,h,i,j] = (strengths-folded r_q) . r_k -> attn region. 16x64 per wave.
__global__ void quantum_scores(const bf16* __restrict__ Rq, const bf16* __restrict__ Rk,
                               float* __restrict__ attn) {
    const int wave = threadIdx.x >> 5;
    const int tile = blockIdx.x * (blockDim.x >> 5) + wave;   // BB*NH*64*16 = 32768
    const int bh  = tile >> 10;
    const int rem = tile & 1023;
    const int ti  = rem >> 4;
    const int tjb = rem & 15;
    const bf16* Ab = Rq + ((size_t)bh * SL + ti * 16) * DKK;
    const v16bf a0 = load_frag_bf16(Ab, DKK, 0);
    const v16bf a1 = load_frag_bf16(Ab, DKK, 32);
    const int lane = threadIdx.x & 31;
    const int nloc = lane & 15;
    const int mbase = (lane >> 4) << 3;
#pragma unroll
    for (int j = 0; j < 4; j++) {
        const int tj = tjb * 4 + j;
        const bf16* Bb = Rk + ((size_t)bh * SL + tj * 16) * DKK;
        v8f acc = {};
        acc = wmma_bf16(a0, load_frag_bf16(Bb, DKK, 0),  acc);
        acc = wmma_bf16(a1, load_frag_bf16(Bb, DKK, 32), acc);
        float* out = attn + (size_t)bh * SL * SL + (size_t)(ti * 16) * SL + tj * 16 + nloc;
#pragma unroll
        for (int v = 0; v < 8; v++) out[(size_t)(mbase + v) * SL] = acc[v];
    }
}

// In-place head mix: final[g] = sum_h q[h] * (delta(h,g) + 0.1*entangle[h,g])
__global__ void head_mix(float* __restrict__ attn, const float* __restrict__ entangle) {
    __shared__ float E[NH * NH];
    if (threadIdx.x < NH * NH) E[threadIdx.x] = entangle[threadIdx.x];
    __syncthreads();
    const size_t idx = (size_t)blockIdx.x * blockDim.x + threadIdx.x;   // BB*S*S
    const int b = (int)(idx >> 20);
    const size_t ij = idx & ((1u << 20) - 1);
    float* base = attn + (size_t)b * NH * SL * SL + ij;
    float q[NH];
#pragma unroll
    for (int h = 0; h < NH; h++) q[h] = base[(size_t)h * SL * SL];
#pragma unroll
    for (int g = 0; g < NH; g++) {
        float s = 0.f;
#pragma unroll
        for (int h = 0; h < NH; h++) s += q[h] * E[h * NH + g];
        base[(size_t)g * SL * SL] = q[g] + 0.1f * s;
    }
}

// attn += 0.125 * Q.K^T, in place over the mixed quantum scores. 16x64 per wave.
__global__ void classical_scores(const bf16* __restrict__ Qb, const bf16* __restrict__ Kb,
                                 float* __restrict__ attn) {
    const int wave = threadIdx.x >> 5;
    const int tile = blockIdx.x * (blockDim.x >> 5) + wave;   // 32768
    const int bh  = tile >> 10;
    const int rem = tile & 1023;
    const int ti  = rem >> 4;
    const int tjb = rem & 15;
    const bf16* Ab = Qb + ((size_t)bh * SL + ti * 16) * DKK;
    const v16bf a0 = load_frag_bf16(Ab, DKK, 0);
    const v16bf a1 = load_frag_bf16(Ab, DKK, 32);
    const int lane = threadIdx.x & 31;
    const int nloc = lane & 15;
    const int mbase = (lane >> 4) << 3;
#pragma unroll
    for (int j = 0; j < 4; j++) {
        const int tj = tjb * 4 + j;
        const bf16* Bb = Kb + ((size_t)bh * SL + tj * 16) * DKK;
        float* out = attn + (size_t)bh * SL * SL + (size_t)(ti * 16) * SL + tj * 16 + nloc;
        float mixed[8];
#pragma unroll
        for (int v = 0; v < 8; v++) mixed[v] = out[(size_t)(mbase + v) * SL];
        v8f acc = {};
        acc = wmma_bf16(a0, load_frag_bf16(Bb, DKK, 0),  acc);
        acc = wmma_bf16(a1, load_frag_bf16(Bb, DKK, 32), acc);
#pragma unroll
        for (int v = 0; v < 8; v++) out[(size_t)(mbase + v) * SL] = mixed[v] + 0.125f * acc[v];
    }
}

// Row softmax over last axis (1024), one 256-thread block per row, in place.
__global__ void softmax_rows(float* __restrict__ attn) {
    __shared__ float red[256];
    float* p = attn + (size_t)blockIdx.x * SL;
    const int tid = threadIdx.x;
    float x0 = p[tid], x1 = p[tid + 256], x2 = p[tid + 512], x3 = p[tid + 768];
    red[tid] = fmaxf(fmaxf(x0, x1), fmaxf(x2, x3));
    __syncthreads();
    for (int s = 128; s > 0; s >>= 1) {
        if (tid < s) red[tid] = fmaxf(red[tid], red[tid + s]);
        __syncthreads();
    }
    const float m = red[0];
    __syncthreads();
    x0 = __expf(x0 - m); x1 = __expf(x1 - m); x2 = __expf(x2 - m); x3 = __expf(x3 - m);
    red[tid] = x0 + x1 + x2 + x3;
    __syncthreads();
    for (int s = 128; s > 0; s >>= 1) {
        if (tid < s) red[tid] += red[tid + s];
        __syncthreads();
    }
    const float inv = 1.0f / red[0];
    p[tid] = x0 * inv; p[tid + 256] = x1 * inv; p[tid + 512] = x2 * inv; p[tid + 768] = x3 * inv;
}

// Oh[b,s,h*64+c] = sum_t attn[b,h,s,t] * V[b,h,t,c]  (V stored transposed [b,h,c,t]).
// Block owns one (b,h) and 8 s-tiles; the 64-row Vt panel is TDM-staged in chunks.
__global__ void attn_v(const float* __restrict__ attn, const bf16* __restrict__ Vt,
                       bf16* __restrict__ Oh) {
    __shared__ __align__(16) bf16 smV[64 * SMLD];
    const int wave = threadIdx.x >> 5;
    const int bh = blockIdx.x >> 3;            // 32 (b,h) pairs
    const int sb = blockIdx.x & 7;             // 8 s-blocks of 128 rows
    const int ti = sb * 8 + wave;              // s tile (16 rows)
    const float* Ab = attn + (size_t)bh * SL * SL + (size_t)(ti * 16) * SL;
    const bf16*  Vg = Vt + (size_t)bh * DKK * SL;
    v8f acc[4] = {};
    for (int ch = 0; ch < 4; ++ch) {
        __syncthreads();
        if (threadIdx.x < 32) {
            tdm_stage_panel((unsigned)(unsigned long long)(void*)smV,
                            (unsigned long long)(const void*)(Vg + ch * 256), SL);
            __builtin_amdgcn_s_wait_tensorcnt(0);
        }
        __syncthreads();
        for (int kk = 0; kk < 256; kk += 32) {
            v16bf a = load_frag_f32(Ab, SL, ch * 256 + kk);
#pragma unroll
            for (int c = 0; c < 4; c++) {
                v16bf b = load_frag_lds(smV, c * 16, kk);
                acc[c] = wmma_bf16(a, b, acc[c]);
            }
        }
    }
    const int lane = threadIdx.x & 31;
    const int nloc = lane & 15;
    const int mbase = (lane >> 4) << 3;
    const int b = bh >> 4, h = bh & (NH - 1);
#pragma unroll
    for (int c = 0; c < 4; c++) {
        const int col = h * DKK + c * 16 + nloc;
#pragma unroll
        for (int v = 0; v < 8; v++) {
            const int s = ti * 16 + mbase + v;
            Oh[((size_t)(b * SL + s)) * DD + col] = (bf16)acc[c][v];
        }
    }
}

// out[m,n] = Oh[m,:] . Wo[n,:] + bo[n]; same TDM-staged structure, f32 output.
__global__ void out_proj(const bf16* __restrict__ Oh, const bf16* __restrict__ Wob,
                         const float* __restrict__ bo, float* __restrict__ out) {
    __shared__ __align__(16) bf16 smB[64 * SMLD];
    const int wave = threadIdx.x >> 5;
    const int bm = blockIdx.x >> 4;
    const int bn = blockIdx.x & 15;
    const bf16* Ab = Oh + ((size_t)bm * 256 + wave * 32) * DD;
    const bf16* Bg = Wob + (size_t)bn * 64 * DD;
    v8f acc[2][4] = {};
    for (int ch = 0; ch < 4; ++ch) {
        __syncthreads();
        if (threadIdx.x < 32) {
            tdm_stage_panel((unsigned)(unsigned long long)(void*)smB,
                            (unsigned long long)(const void*)(Bg + ch * 256), DD);
            __builtin_amdgcn_s_wait_tensorcnt(0);
        }
        __syncthreads();
        for (int kk = 0; kk < 256; kk += 32) {
            const int k0 = ch * 256 + kk;
            __builtin_prefetch(Ab + ((threadIdx.x & 15) * DD) + k0 + 64, 0, 0);
            v16bf a0 = load_frag_bf16(Ab, DD, k0);
            v16bf a1 = load_frag_bf16(Ab + 16 * DD, DD, k0);
#pragma unroll
            for (int j = 0; j < 4; j++) {
                v16bf b = load_frag_lds(smB, j * 16, kk);
                acc[0][j] = wmma_bf16(a0, b, acc[0][j]);
                acc[1][j] = wmma_bf16(a1, b, acc[1][j]);
            }
        }
    }
    const int lane = threadIdx.x & 31;
    const int nloc = lane & 15;
    const int mbase = (lane >> 4) << 3;
#pragma unroll
    for (int ms = 0; ms < 2; ms++) {
#pragma unroll
        for (int j = 0; j < 4; j++) {
            const int ncol = bn * 64 + j * 16 + nloc;
            const float bvv = bo[ncol];
#pragma unroll
            for (int v = 0; v < 8; v++)
                out[(size_t)(bm * 256 + wave * 32 + ms * 16 + mbase + v) * DD + ncol] =
                    acc[ms][j][v] + bvv;
        }
    }
}

// ---------------- host launcher ----------------
extern "C" void kernel_launch(void* const* d_in, const int* in_sizes, int n_in,
                              void* d_out, int out_size, void* d_ws, size_t ws_size,
                              hipStream_t stream) {
    const float* x   = (const float*)d_in[0];
    const float* Wq  = (const float*)d_in[1];
    const float* bq  = (const float*)d_in[2];
    const float* Wk  = (const float*)d_in[3];
    const float* bk  = (const float*)d_in[4];
    const float* Wv  = (const float*)d_in[5];
    const float* bv  = (const float*)d_in[6];
    const float* Wo  = (const float*)d_in[7];
    const float* bo  = (const float*)d_in[8];
    const float* Wt  = (const float*)d_in[9];
    const float* bt  = (const float*)d_in[10];
    const float* ent = (const float*)d_in[11];
    const float* str = (const float*)d_in[12];

    float* out  = (float*)d_out;                           // [2,1024,1024]
    float* attn = out + (size_t)BB * SL * DD;              // [2,16,1024,1024]

    char* w = (char*)d_ws;
    bf16* Xb  = (bf16*)w; w += (size_t)MTOT * DD * 2;      // 4 MB
    bf16* Wqb = (bf16*)w; w += (size_t)DD * DD * 2;        // 2 MB
    bf16* Wkb = (bf16*)w; w += (size_t)DD * DD * 2;
    bf16* Wvb = (bf16*)w; w += (size_t)DD * DD * 2;
    bf16* Wob = (bf16*)w; w += (size_t)DD * DD * 2;
    bf16* Qb  = (bf16*)w; w += (size_t)BHS * DKK * 2;      // [b,h,s,dk]
    bf16* Kb  = (bf16*)w; w += (size_t)BHS * DKK * 2;
    bf16* Vt  = (bf16*)w; w += (size_t)BHS * DKK * 2;      // [b,h,dk,s]
    bf16* Rq  = (bf16*)w; w += (size_t)BHS * DKK * 2;
    bf16* Rk  = (bf16*)w; w += (size_t)BHS * DKK * 2;
    bf16* Oh  = (bf16*)w; w += (size_t)MTOT * DD * 2;

    const int T = 256;
    // 1) fp32 -> bf16 conversions
    cvt_f32_bf16<<<(MTOT * DD + T - 1) / T, T, 0, stream>>>(x,  Xb,  MTOT * DD);
    cvt_f32_bf16<<<(DD * DD + T - 1) / T,   T, 0, stream>>>(Wq, Wqb, DD * DD);
    cvt_f32_bf16<<<(DD * DD + T - 1) / T,   T, 0, stream>>>(Wk, Wkb, DD * DD);
    cvt_f32_bf16<<<(DD * DD + T - 1) / T,   T, 0, stream>>>(Wv, Wvb, DD * DD);
    cvt_f32_bf16<<<(DD * DD + T - 1) / T,   T, 0, stream>>>(Wo, Wob, DD * DD);
    // 2) QKV projections (WMMA + TDM-staged weight panels)
    proj_qkv<<<128, T, 0, stream>>>(Xb, Wqb, bq, Qb, 0);
    proj_qkv<<<128, T, 0, stream>>>(Xb, Wkb, bk, Kb, 0);
    proj_qkv<<<128, T, 0, stream>>>(Xb, Wvb, bv, Vt, 1);
    // 3) theta projection -> r (strengths folded into r_q)
    theta_r<<<(BHS * DKK + T - 1) / T, T, 0, stream>>>(Qb, Rq, Wt, bt, str, 1);
    theta_r<<<(BHS * DKK + T - 1) / T, T, 0, stream>>>(Kb, Rk, Wt, bt, str, 0);
    // 4) quantum scores (WMMA) straight into d_out attn region
    quantum_scores<<<4096, T, 0, stream>>>(Rq, Rk, attn);
    // 5) in-place cross-head entangle mix (delta + 0.1*E)
    head_mix<<<(BB * SL * SL) / T, T, 0, stream>>>(attn, ent);
    // 6) add classical QK^T/8 in place (WMMA)
    classical_scores<<<4096, T, 0, stream>>>(Qb, Kb, attn);
    // 7) softmax per row (in place -> final attn output)
    softmax_rows<<<BB * NH * SL, T, 0, stream>>>(attn);
    // 8) attn @ V (WMMA + TDM-staged V panels)
    attn_v<<<256, T, 0, stream>>>(attn, Vt, Oh);
    // 9) output projection + bias (WMMA + TDM-staged Wo panels)
    out_proj<<<128, T, 0, stream>>>(Oh, Wob, bo, out);
}
